// GCN_14353780703430
// MI455X (gfx1250) — compile-verified, matrix operand order checked by reading, not run on hardware
//
#include <hip/hip_runtime.h>
#include <hip/hip_bf16.h>

#define IN_DIM 512
#define HID    128
#define OUT_DIM 64

typedef _Float16 v16h __attribute__((ext_vector_type(16)));
typedef _Float16 h8_t __attribute__((ext_vector_type(8)));
typedef float    v8f  __attribute__((ext_vector_type(8)));

// Native f32 atomic add (global_atomic_add_f32, no CAS loop).
__device__ __forceinline__ void atomic_add_f32(float* p, float v) {
  unsafeAtomicAdd(p, v);
}

// ---------------- weight prep: f32 [K,C] -> f16 transposed [C,K] ----------------
__global__ void wprep_kernel(const float* __restrict__ W, _Float16* __restrict__ Wt,
                             int K, int C) {
  int idx = blockIdx.x * blockDim.x + threadIdx.x;
  if (idx >= K * C) return;
  int k = idx / C, c = idx % C;
  Wt[(size_t)c * K + k] = (_Float16)W[idx];
}

// ---------------- degree / norm ----------------
__global__ void fill1_kernel(float* __restrict__ p, int n) {
  int i = blockIdx.x * blockDim.x + threadIdx.x;
  if (i < n) p[i] = 1.0f;   // self-loop contributes 1 to every node's degree
}

__global__ void deg_kernel(const long long* __restrict__ ei, float* __restrict__ deg, int nE) {
  int e = blockIdx.x * blockDim.x + threadIdx.x;
  if (e < nE) atomic_add_f32(&deg[(int)ei[(size_t)nE + e]], 1.0f);
}

__global__ void dinv_kernel(const float* __restrict__ deg, float* __restrict__ dinv, int n) {
  int i = blockIdx.x * blockDim.x + threadIdx.x;
  if (i >= n) return;
  float d = deg[i];
  dinv[i] = d > 0.0f ? __frsqrt_rn(d) : 0.0f;
}

__global__ void norm_kernel(const long long* __restrict__ ei, const float* __restrict__ dinv,
                            float* __restrict__ norm, int nE, int nTot) {
  int e = blockIdx.x * blockDim.x + threadIdx.x;
  if (e >= nTot) return;
  float v;
  if (e < nE) {
    v = dinv[(int)ei[e]] * dinv[(int)ei[(size_t)nE + e]];
  } else {
    float d = dinv[e - nE];   // self loop
    v = d * d;
  }
  norm[e] = v;
}

// ---------------- WMMA GEMM: D[M,C] = A[M,K] @ Bt[C,K]^T ----------------
// One wave computes one 16x16 tile of D via v_wmma_f32_16x16x32_f16.
// A fragment layout (16-bit A 16x32, ISA 7.12.2): lane l holds row (l&15),
// K-elements {kb..kb+7, kb+16..kb+23} with kb = (l>>4)*8. B fragment mirrors it
// using the transposed weights (column c of W == row c of Wt, contiguous in K).
template <bool A_IS_F32>
__global__ void gemm_wmma_kernel(const void* __restrict__ Av,
                                 const _Float16* __restrict__ Bt,
                                 float* __restrict__ D,
                                 int K, int C, int mtiles) {
  int lane = threadIdx.x & 31;
  int wave = threadIdx.x >> 5;
  int mtile = blockIdx.x * 8 + wave;
  if (mtile >= mtiles) return;                 // wave-uniform exit
  int row0 = mtile * 16;
  int col0 = blockIdx.y * 16;
  int m  = lane & 15;
  int kb = (lane >> 4) * 8;

  const _Float16* A16 = (const _Float16*)Av;
  const float*    A32 = (const float*)Av;
  const _Float16* bp  = Bt + (size_t)(col0 + m) * K + kb;

  v8f acc = {};
  for (int k = 0; k < K; k += 32) {
    v16h a, b;
    if (A_IS_F32) {
      const float* ap = A32 + (size_t)(row0 + m) * K + k + kb;
      if (k + 32 < K) __builtin_prefetch(ap + 32, 0, 0);   // global_prefetch_b8
#pragma unroll
      for (int i = 0; i < 8; ++i) { a[i] = (_Float16)ap[i]; a[8 + i] = (_Float16)ap[16 + i]; }
    } else {
      const _Float16* ap = A16 + (size_t)(row0 + m) * K + k + kb;
      if (k + 32 < K) __builtin_prefetch(ap + 32, 0, 0);
      h8_t lo = *(const h8_t*)ap;
      h8_t hi = *(const h8_t*)(ap + 16);
#pragma unroll
      for (int i = 0; i < 8; ++i) { a[i] = lo[i]; a[8 + i] = hi[i]; }
    }
    {
      h8_t lo = *(const h8_t*)(bp + k);
      h8_t hi = *(const h8_t*)(bp + k + 16);
#pragma unroll
      for (int i = 0; i < 8; ++i) { b[i] = lo[i]; b[8 + i] = hi[i]; }
    }
    acc = __builtin_amdgcn_wmma_f32_16x16x32_f16(false, a, false, b,
                                                 (short)0, acc, false, false);
  }

  // D layout: vgpr r -> row r + (lane>=16 ? 8 : 0), col (lane&15)
  int rbase = row0 + ((lane >> 4) * 8);
  int col   = col0 + (lane & 15);
#pragma unroll
  for (int r = 0; r < 8; ++r)
    D[(size_t)(rbase + r) * C + col] = acc[r];
}

// ---------------- scatter-aggregate: agg[dst] += norm * h[src] ----------------
// One wave per edge; each lane owns a contiguous C/32 chunk -> one b128/b64
// gather per lane + native f32 atomic scatter. h and agg stay L2-resident.
template <int C>
__global__ void aggregate_kernel(const float* __restrict__ h,
                                 const long long* __restrict__ ei,
                                 const float* __restrict__ norm,
                                 float* __restrict__ agg,
                                 int nE, int nN) {
  constexpr int PER = C / 32;                     // 4 (C=128) or 2 (C=64)
  typedef float vec_t __attribute__((ext_vector_type(PER)));
  int e = (blockIdx.x * blockDim.x + threadIdx.x) >> 5;
  int lane = threadIdx.x & 31;
  int nTot = nE + nN;
  if (e >= nTot) return;
  int s, d;
  if (e < nE) { s = (int)ei[e]; d = (int)ei[(size_t)nE + e]; }
  else        { s = d = e - nE; }
  float w = norm[e];
  const float* hp = h   + (size_t)s * C + lane * PER;
  float*       op = agg + (size_t)d * C + lane * PER;
  vec_t hv = *(const vec_t*)hp;
#pragma unroll
  for (int i = 0; i < PER; ++i)
    atomic_add_f32(&op[i], w * hv[i]);
}

// ---------------- bias + ReLU -> f16 activations ----------------
__global__ void bias_relu_kernel(const float* __restrict__ agg, const float* __restrict__ b,
                                 _Float16* __restrict__ ah, int n, int C) {
  int i = blockIdx.x * blockDim.x + threadIdx.x;
  if (i >= n) return;
  float v = agg[i] + b[i % C];
  ah[i] = (_Float16)fmaxf(v, 0.0f);
}

// ---------------- final bias + log_softmax (C = 64, one wave per node) ----------------
__global__ void logsoftmax_kernel(const float* __restrict__ agg, const float* __restrict__ b,
                                  float* __restrict__ out, int nN) {
  int node = (blockIdx.x * blockDim.x + threadIdx.x) >> 5;
  int lane = threadIdx.x & 31;
  if (node >= nN) return;
  const float* ap = agg + (size_t)node * OUT_DIM;
  float v0 = ap[lane]      + b[lane];
  float v1 = ap[lane + 32] + b[lane + 32];
  float m = fmaxf(v0, v1);
#pragma unroll
  for (int o = 16; o > 0; o >>= 1) m = fmaxf(m, __shfl_xor(m, o, 32));
  float s = __expf(v0 - m) + __expf(v1 - m);
#pragma unroll
  for (int o = 16; o > 0; o >>= 1) s += __shfl_xor(s, o, 32);
  float lse = m + __logf(s);
  float* op = out + (size_t)node * OUT_DIM;
  op[lane]      = v0 - lse;
  op[lane + 32] = v1 - lse;
}

// ---------------- launcher ----------------
extern "C" void kernel_launch(void* const* d_in, const int* in_sizes, int n_in,
                              void* d_out, int out_size, void* d_ws, size_t ws_size,
                              hipStream_t stream) {
  (void)n_in; (void)out_size; (void)ws_size;
  const float*     x  = (const float*)d_in[0];
  const long long* ei = (const long long*)d_in[1];   // int64 edge_index [2, E]
  const float* W1 = (const float*)d_in[2]; const float* b1 = (const float*)d_in[3];
  const float* W2 = (const float*)d_in[4]; const float* b2 = (const float*)d_in[5];
  const float* W3 = (const float*)d_in[6]; const float* b3 = (const float*)d_in[7];
  const float* W4 = (const float*)d_in[8]; const float* b4 = (const float*)d_in[9];

  const int nN   = in_sizes[0] / IN_DIM;   // 100000
  const int nE   = in_sizes[1] / 2;        // 1600000
  const int nTot = nE + nN;

  // workspace carve-out (~136 MB)
  char* ws = (char*)d_ws;
  size_t off = 0;
  auto carve = [&](size_t bytes) -> char* {
    char* p = ws + off;
    off = (off + bytes + 255) & ~(size_t)255;
    return p;
  };
  float*    h    = (float*)carve((size_t)nN * HID * 4);
  float*    agg  = (float*)carve((size_t)nN * HID * 4);
  _Float16* ah   = (_Float16*)carve((size_t)nN * HID * 2);
  float*    norm = (float*)carve((size_t)nTot * 4);
  float*    dinv = (float*)carve((size_t)nN * 4);
  float*    deg  = (float*)carve((size_t)nN * 4);
  _Float16* Wt1  = (_Float16*)carve((size_t)IN_DIM * HID * 2);
  _Float16* Wt2  = (_Float16*)carve((size_t)HID * HID * 2);
  _Float16* Wt3  = (_Float16*)carve((size_t)HID * HID * 2);
  _Float16* Wt4  = (_Float16*)carve((size_t)HID * OUT_DIM * 2);

  // ---- weight prep (f16, transposed) ----
  wprep_kernel<<<(IN_DIM * HID + 255) / 256, 256, 0, stream>>>(W1, Wt1, IN_DIM, HID);
  wprep_kernel<<<(HID * HID + 255) / 256, 256, 0, stream>>>(W2, Wt2, HID, HID);
  wprep_kernel<<<(HID * HID + 255) / 256, 256, 0, stream>>>(W3, Wt3, HID, HID);
  wprep_kernel<<<(HID * OUT_DIM + 255) / 256, 256, 0, stream>>>(W4, Wt4, HID, OUT_DIM);

  // ---- symmetric normalization ----
  fill1_kernel<<<(nN + 255) / 256, 256, 0, stream>>>(deg, nN);
  deg_kernel<<<(nE + 255) / 256, 256, 0, stream>>>(ei, deg, nE);
  dinv_kernel<<<(nN + 255) / 256, 256, 0, stream>>>(deg, dinv, nN);
  norm_kernel<<<(nTot + 255) / 256, 256, 0, stream>>>(ei, dinv, norm, nE, nTot);

  const int mtiles = (nN + 15) / 16;            // 6250, exact
  const dim3 gemmGridH((mtiles + 7) / 8, HID / 16);
  const dim3 gemmGridO((mtiles + 7) / 8, OUT_DIM / 16);
  const int aggBlocks = (nTot + 7) / 8;         // one wave per edge, 8 waves/block
  const int ewBlocks  = ((size_t)nN * HID + 255) / 256;

  // ---- layer 1 (A = f32 x) ----
  gemm_wmma_kernel<true><<<gemmGridH, 256, 0, stream>>>((const void*)x, Wt1, h, IN_DIM, HID, mtiles);
  hipMemsetAsync(agg, 0, (size_t)nN * HID * 4, stream);
  aggregate_kernel<HID><<<aggBlocks, 256, 0, stream>>>(h, ei, norm, agg, nE, nN);
  bias_relu_kernel<<<ewBlocks, 256, 0, stream>>>(agg, b1, ah, nN * HID, HID);

  // ---- layer 2 ----
  gemm_wmma_kernel<false><<<gemmGridH, 256, 0, stream>>>((const void*)ah, Wt2, h, HID, HID, mtiles);
  hipMemsetAsync(agg, 0, (size_t)nN * HID * 4, stream);
  aggregate_kernel<HID><<<aggBlocks, 256, 0, stream>>>(h, ei, norm, agg, nE, nN);
  bias_relu_kernel<<<ewBlocks, 256, 0, stream>>>(agg, b2, ah, nN * HID, HID);

  // ---- layer 3 ----
  gemm_wmma_kernel<false><<<gemmGridH, 256, 0, stream>>>((const void*)ah, Wt3, h, HID, HID, mtiles);
  hipMemsetAsync(agg, 0, (size_t)nN * HID * 4, stream);
  aggregate_kernel<HID><<<aggBlocks, 256, 0, stream>>>(h, ei, norm, agg, nE, nN);
  bias_relu_kernel<<<ewBlocks, 256, 0, stream>>>(agg, b3, ah, nN * HID, HID);

  // ---- layer 4 + log_softmax ----
  gemm_wmma_kernel<false><<<gemmGridO, 256, 0, stream>>>((const void*)ah, Wt4, h, HID, OUT_DIM, mtiles);
  hipMemsetAsync(agg, 0, (size_t)nN * OUT_DIM * 4, stream);
  aggregate_kernel<OUT_DIM><<<aggBlocks, 256, 0, stream>>>(h, ei, norm, agg, nE, nN);
  logsoftmax_kernel<<<(nN + 7) / 8, 256, 0, stream>>>(agg, b4, (float*)d_out, nN);
}
